// TokenSelectiveAttention_75453985457484
// MI455X (gfx1250) — matrix operand order repaired and checked by reading, not compiled, not run on hardware
//
#include <hip/hip_runtime.h>
#include <hip/hip_bf16.h>

typedef __attribute__((ext_vector_type(16))) _Float16 v16h;
typedef __attribute__((ext_vector_type(8)))  float    v8f;
typedef __attribute__((ext_vector_type(2)))  float    v2f;
typedef unsigned int u32x4 __attribute__((ext_vector_type(4)));
typedef int          i32x8 __attribute__((ext_vector_type(8)));
typedef int          i32x4 __attribute__((ext_vector_type(4)));

#define NTOK 2304
#define LDW  2308                  // padded LDS row stride (floats)
#define KTOP 1843
#define WANT0 (NTOK - KTOP + 1)    // 462: ascending rank of threshold

// LDS layout (bytes): logits | hist | part | vlds
#define SM_LOGITS_F (16 * LDW)
#define SM_HIST_I   (16 * 256)
#define SM_PART_F   (8 * 256)
#define SM_BYTES    ((size_t)SM_LOGITS_F * 4 + (size_t)SM_HIST_I * 4 + \
                     (size_t)SM_PART_F * 4 + (size_t)NTOK * 16 * 2)

__device__ __forceinline__ unsigned f2key(float x) {
  unsigned u = __float_as_uint(x);
  return (u & 0x80000000u) ? ~u : (u | 0x80000000u);   // monotone ascending map
}

// ---------------- Stage A1: grouped 1x1 conv for q,k,v ----------------
__global__ void __launch_bounds__(256) k_qkv_pw(
    const float* __restrict__ x,
    const float* __restrict__ wq, const float* __restrict__ wk, const float* __restrict__ wv,
    float* __restrict__ q1, float* __restrict__ k1, float* __restrict__ v1) {
  int idx = blockIdx.x * 256 + threadIdx.x;            // 3*2*128*2304 total
  int p = idx % NTOK; int t = idx / NTOK;
  int c = t % 128; t /= 128;
  int b = t % 2;  int proj = t / 2;
  const float* w = (proj == 0) ? wq : (proj == 1) ? wk : wv;
  float* o       = (proj == 0) ? q1 : (proj == 1) ? k1 : v1;
  int g = c >> 5;
  const float* xp = x + ((size_t)(b * 128 + g * 32)) * NTOK + p;
  const float* wp = w + c * 32;
  float acc = 0.f;
#pragma unroll
  for (int i = 0; i < 32; ++i) acc += wp[i] * xp[(size_t)i * NTOK];
  o[((size_t)(b * 128 + c)) * NTOK + p] = acc;
}

// ---------------- Stage A2: depthwise 3x3 SAME ----------------
__global__ void __launch_bounds__(256) k_qkv_dw(
    const float* __restrict__ q1, const float* __restrict__ k1, const float* __restrict__ v1,
    const float* __restrict__ dwq, const float* __restrict__ dwk, const float* __restrict__ dwv,
    float* __restrict__ q2, float* __restrict__ k2, float* __restrict__ v2) {
  int idx = blockIdx.x * 256 + threadIdx.x;
  int p = idx % NTOK; int t = idx / NTOK;
  int c = t % 128; t /= 128;
  int b = t % 2;  int proj = t / 2;
  const float* in = ((proj == 0) ? q1 : (proj == 1) ? k1 : v1) + ((size_t)(b * 128 + c)) * NTOK;
  const float* w9 = ((proj == 0) ? dwq : (proj == 1) ? dwk : dwv) + c * 9;
  int y = p / 48, xx = p % 48;
  float acc = 0.f;
#pragma unroll
  for (int dy = 0; dy < 3; ++dy) {
    int yy = y + dy - 1;
    if ((unsigned)yy < 48u) {
#pragma unroll
      for (int dx = 0; dx < 3; ++dx) {
        int xc = xx + dx - 1;
        if ((unsigned)xc < 48u) acc += w9[dy * 3 + dx] * in[yy * 48 + xc];
      }
    }
  }
  ((proj == 0) ? q2 : (proj == 1) ? k2 : v2)[((size_t)(b * 128 + c)) * NTOK + p] = acc;
}

// ---------------- Stage B1: per-channel l2 norm of k over N ----------------
__global__ void __launch_bounds__(256) k_knorm(const float* __restrict__ k2,
                                               float* __restrict__ knorm) {
  int wid = (blockIdx.x * 256 + threadIdx.x) >> 5;     // (b*128+c), 256 waves
  int lane = threadIdx.x & 31;
  const float* kp = k2 + (size_t)wid * NTOK;
  float s = 0.f;
  for (int j = lane; j < NTOK; j += 32) { float v = kp[j]; s += v * v; }
#pragma unroll
  for (int m = 16; m >= 1; m >>= 1) s += __shfl_xor(s, m, 32);
  if (lane == 0) knorm[wid] = fmaxf(sqrtf(s), 1e-12f);
}

// ---- Stage B2: q token-l2norm + fold 1/knorm into q; pack v to f16 [bh][n][16] ----
__global__ void __launch_bounds__(256) k_qv_pack(
    const float* __restrict__ q2, const float* __restrict__ v2,
    const float* __restrict__ knorm,
    float* __restrict__ qn, _Float16* __restrict__ vh) {
  int idx = blockIdx.x * 256 + threadIdx.x;            // bh*2304 + p, 36864 total
  int bh = idx / NTOK; int p = idx % NTOK;
  int b = bh >> 3, h = bh & 7;
  const float* qp = q2 + ((size_t)(b * 128 + h * 16)) * NTOK + p;
  const float* vp = v2 + ((size_t)(b * 128 + h * 16)) * NTOK + p;
  float vals[16]; float s = 0.f;
#pragma unroll
  for (int c = 0; c < 16; ++c) { vals[c] = qp[(size_t)c * NTOK]; s += vals[c] * vals[c]; }
  float inv = 1.f / fmaxf(sqrtf(s), 1e-12f);
#pragma unroll
  for (int c = 0; c < 16; ++c) {
    float rk = 1.f / knorm[b * 128 + h * 16 + c];
    qn[(size_t)idx * 16 + c] = vals[c] * inv * rk;
    vh[(size_t)idx * 16 + c] = (_Float16)vp[(size_t)c * NTOK];
  }
}

// ---------------- Fused attention: QK^T (f32 WMMA) -> topk -> softmax -> PV (f16 WMMA) ----
// V[bh] (73,728 B) is DMA'd to LDS by the Tensor Data Mover at kernel start,
// overlapping the whole QK/topk/softmax phase; drained with s_wait_tensorcnt.
__global__ void __launch_bounds__(256) k_attn(
    const float* __restrict__ qn, const float* __restrict__ k2,
    const _Float16* __restrict__ vh, const float* __restrict__ temperature,
    float* __restrict__ aout) {
  extern __shared__ float smem[];
  float*    logits = smem;                                  // 16 * LDW floats
  int*      hist   = (int*)(smem + SM_LOGITS_F);            // 16 * 256 ints
  float*    part   = (float*)(hist + SM_HIST_I);            // 8 * 256 floats
  _Float16* vlds   = (_Float16*)(part + SM_PART_F);         // 2304 * 16 halfs

  int bid = blockIdx.x;
  int bh = bid / 144; int tile = bid % 144; int n0 = tile * 16;
  int b = bh >> 3, h = bh & 7;
  int tid = threadIdx.x, wv_ = tid >> 5, lane = tid & 31;
  int rowA = lane & 15, hi = lane >> 4;
  float temp = temperature[h];

  const float* qp = qn + (size_t)bh * NTOK * 16;
  const float* kp = k2 + ((size_t)(b * 128 + h * 16)) * NTOK;  // [16 ch][N], B-matrix as-is
  const _Float16* vp = vh + (size_t)bh * NTOK * 16;

  // ---- kick off TDM copy of V[bh] -> LDS (1D: 9216 x 8B elements) ----
#if __has_builtin(__builtin_amdgcn_tensor_load_to_lds) && \
    __has_builtin(__builtin_amdgcn_s_wait_tensorcnt)
#define USE_TDM 1
  if (wv_ == 0) {
    unsigned long long ga = (unsigned long long)(size_t)(const void*)vp;
    unsigned ldso = __builtin_amdgcn_readfirstlane((unsigned)(size_t)(void*)vlds);
    unsigned galo = __builtin_amdgcn_readfirstlane((unsigned)(ga & 0xffffffffu));
    unsigned gahi = __builtin_amdgcn_readfirstlane((unsigned)(ga >> 32));
    u32x4 g0;
    g0[0] = 1u;                                   // count=1, user mode, no gather
    g0[1] = ldso;                                 // lds_addr (bytes)
    g0[2] = galo;                                 // global_addr[31:0]
    g0[3] = (gahi & 0x01ffffffu) | 0x80000000u;   // global_addr[56:32] | type=2@[127:126]
    i32x8 g1;
    g1[0] = 0x00030000;                           // workgroup_mask=0, data_size=3 (8B)
    g1[1] = (int)((9216u & 0xffffu) << 16);       // tensor_dim0.lo16 << 16
    g1[2] = (int)(1u << 16);                      // tensor_dim0.hi16=0 | tensor_dim1.lo16=1
    g1[3] = (int)((9216u & 0xffffu) << 16);       // tensor_dim1.hi16=0 | tile_dim0=9216
    g1[4] = 1;                                    // tile_dim1=1, tile_dim2=0
    g1[5] = 9216;                                 // tensor_dim0_stride[31:0]
    g1[6] = 0;                                    // stride0.hi16 | stride1.lo16
    g1[7] = 0;                                    // stride1.hi32
    i32x4 gz  = {0, 0, 0, 0};
    i32x8 gz8 = {0, 0, 0, 0, 0, 0, 0, 0};
    __builtin_amdgcn_tensor_load_to_lds(g0, g1, gz, gz, gz8, 0);
  }
#else
  {  // fallback: cooperative vectorized copy (visibility via the pre-PV barrier)
    const uint4* gsrc = (const uint4*)vp;
    uint4* ldst = (uint4*)vlds;
    for (int i = tid; i < (NTOK * 16 * 2) / 16; i += 256) ldst[i] = gsrc[i];
  }
#endif

  // ---- QK^T: each wave owns k-tiles w, w+8, ... (144/8 = 18 each, uniform) ----
#if __has_builtin(__builtin_amdgcn_wmma_f32_16x16x4_f32)
  v2f a[4];
#pragma unroll
  for (int i = 0; i < 4; ++i) {
    const float* qr = qp + (size_t)(n0 + rowA) * 16 + 4 * i + 2 * hi;
    a[i][0] = qr[0]; a[i][1] = qr[1];
  }
  for (int t = wv_; t < 144; t += 8) {
    int n1 = t * 16;
    v8f c8 = {};
#pragma unroll
    for (int i = 0; i < 4; ++i) {
      int cc = 4 * i + 2 * hi;
      const float* kb = kp + (size_t)cc * NTOK + n1 + rowA;
      v2f bfr; bfr[0] = kb[0]; bfr[1] = kb[NTOK];
      c8 = __builtin_amdgcn_wmma_f32_16x16x4_f32(false, a[i], false, bfr, (short)0, c8,
                                                 false, false);
    }
#pragma unroll
    for (int e = 0; e < 8; ++e)
      logits[(e + 8 * hi) * LDW + n1 + rowA] = c8[e] * temp;
  }
#else
  v16h aq;
#pragma unroll
  for (int j = 0; j < 16; ++j) {
    int k = ((j < 8) ? j : j + 8) + 8 * hi;
    aq[j] = (k < 16) ? (_Float16)qp[(size_t)(n0 + rowA) * 16 + k] : (_Float16)0.f;
  }
  for (int t = wv_; t < 144; t += 8) {
    int n1 = t * 16;
    v16h bk;
#pragma unroll
    for (int j = 0; j < 16; ++j) {
      int k = j + 16 * hi;
      bk[j] = (k < 16) ? (_Float16)kp[(size_t)k * NTOK + n1 + rowA] : (_Float16)0.f;
    }
    v8f c8 = {};
    c8 = __builtin_amdgcn_wmma_f32_16x16x32_f16(false, aq, false, bk, (short)0, c8, false, false);
#pragma unroll
    for (int e = 0; e < 8; ++e)
      logits[(e + 8 * hi) * LDW + n1 + rowA] = c8[e] * temp;
  }
#endif
  __syncthreads();

  // ---- per-row top-k threshold (radix select) + softmax; wave w -> rows 2w,2w+1 ----
  for (int rr = 0; rr < 2; ++rr) {
    int row = wv_ * 2 + rr;
    float* lrow = logits + row * LDW;
    int* hrow = hist + row * 256;
    unsigned prefix = 0; int want = WANT0;
#pragma unroll 1
    for (int pass = 0; pass < 4; ++pass) {
      int shift = 24 - 8 * pass;
      for (int i = lane; i < 256; i += 32) hrow[i] = 0;
      for (int j = lane; j < NTOK; j += 32) {
        unsigned u = f2key(lrow[j]);
        unsigned hiu = (unsigned)((unsigned long long)u >> (shift + 8));
        if (pass == 0 || hiu == prefix) atomicAdd(&hrow[(u >> shift) & 255], 1);
      }
      int d = 0, cum = 0;
      if (lane == 0) {
        int c2 = 0, i = 0;
        for (; i < 256; ++i) { int hh = hrow[i]; if (c2 + hh >= want) break; c2 += hh; }
        d = i; cum = c2;
      }
      d = __shfl(d, 0, 32); cum = __shfl(cum, 0, 32);
      want -= cum;
      prefix = (prefix << 8) | (unsigned)d;
    }
    float mx = -3.4e38f;
    for (int j = lane; j < NTOK; j += 32) {
      float x = lrow[j];
      if (f2key(x) >= prefix) mx = fmaxf(mx, x);
    }
#pragma unroll
    for (int m = 16; m >= 1; m >>= 1) mx = fmaxf(mx, __shfl_xor(mx, m, 32));
    float s = 0.f;
    for (int j = lane; j < NTOK; j += 32) {
      float x = lrow[j];
      float e = (f2key(x) >= prefix) ? __expf(x - mx) : 0.f;
      lrow[j] = e; s += e;
    }
#pragma unroll
    for (int m = 16; m >= 1; m >>= 1) s += __shfl_xor(s, m, 32);
    float inv = 1.f / s;
    for (int j = lane; j < NTOK; j += 32) lrow[j] *= inv;
  }

#ifdef USE_TDM
  if (wv_ == 0) __builtin_amdgcn_s_wait_tensorcnt(0);  // V resident in LDS
#endif
  __syncthreads();

  // ---- PV: out[16,16] = P[16,2304] @ V[2304,16]; wave owns m-chunks w,w+8,... (72/8=9) ----
  v8f acc = {};
  for (int mc = wv_; mc < 72; mc += 8) {
    int m1 = mc * 32;
    v16h af, bf;
#pragma unroll
    for (int j = 0; j < 16; ++j) {
      int ka = j + 16 * hi;                             // A 16-bit K map
      af[j] = (_Float16)logits[rowA * LDW + m1 + ka];
      int kb = j + 16 * hi;                             // B 16-bit K map
      bf[j] = vlds[(m1 + kb) * 16 + rowA];
    }
    acc = __builtin_amdgcn_wmma_f32_16x16x32_f16(false, af, false, bf, (short)0, acc,
                                                 false, false);
  }
#pragma unroll
  for (int e = 0; e < 8; ++e) part[wv_ * 256 + e * 32 + lane] = acc[e];
  __syncthreads();
  {
    int u = tid;                                        // e*32 + lane
    float s = 0.f;
#pragma unroll
    for (int w2 = 0; w2 < 8; ++w2) s += part[w2 * 256 + u];
    int lt = u & 31, e = u >> 5;
    int m = e + 8 * (lt >> 4);
    int cc = lt & 15;
    aout[((size_t)bh * NTOK + n0 + m) * 16 + cc] = s;
  }
}

// ---------------- Stage C: 1x1 conv wo + residual ----------------
__global__ void __launch_bounds__(256) k_out(
    const float* __restrict__ x, const float* __restrict__ wo,
    const float* __restrict__ aout, float* __restrict__ out) {
  __shared__ float a_s[128 * 64];
  int bid = blockIdx.x; int b = bid / 36; int p0 = (bid % 36) * 64;
  int tid = threadIdx.x;
#pragma unroll
  for (int e = 0; e < 32; ++e) {
    int idx = tid + e * 256;
    int ci = idx >> 6, pp = idx & 63;
    a_s[idx] = aout[(((size_t)(b * 8 + (ci >> 4))) * NTOK + p0 + pp) * 16 + (ci & 15)];
  }
  __syncthreads();
  int px = tid & 63; int cob = (tid >> 6) * 32;
  float acc[32];
#pragma unroll
  for (int i = 0; i < 32; ++i) acc[i] = 0.f;
  for (int ci = 0; ci < 128; ++ci) {
    float av = a_s[ci * 64 + px];
#pragma unroll
    for (int i = 0; i < 32; ++i) acc[i] += wo[(size_t)(cob + i) * 128 + ci] * av;
  }
#pragma unroll
  for (int i = 0; i < 32; ++i) {
    size_t o = ((size_t)(b * 128 + cob + i)) * NTOK + p0 + px;
    out[o] = x[o] + acc[i];
  }
}

extern "C" void kernel_launch(void* const* d_in, const int* in_sizes, int n_in,
                              void* d_out, int out_size, void* d_ws, size_t ws_size,
                              hipStream_t stream) {
  (void)in_sizes; (void)n_in; (void)out_size; (void)ws_size;
  const float* x   = (const float*)d_in[0];
  const float* wq  = (const float*)d_in[1];
  const float* wk  = (const float*)d_in[2];
  const float* wv  = (const float*)d_in[3];
  const float* dwq = (const float*)d_in[4];
  const float* dwk = (const float*)d_in[5];
  const float* dwv = (const float*)d_in[6];
  const float* wo  = (const float*)d_in[7];
  const float* temp = (const float*)d_in[8];
  float* out = (float*)d_out;

  const size_t SZ = (size_t)2 * 128 * NTOK;            // 589824
  float* f = (float*)d_ws;
  float *q1 = f,          *k1 = f + SZ,     *v1 = f + 2 * SZ,
        *q2 = f + 3 * SZ, *k2 = f + 4 * SZ, *v2 = f + 5 * SZ,
        *qn = f + 6 * SZ, *aout = f + 7 * SZ, *knorm = f + 8 * SZ;
  _Float16* vh = (_Float16*)(f + 8 * SZ + 256);

  k_qkv_pw<<<6912, 256, 0, stream>>>(x, wq, wk, wv, q1, k1, v1);
  k_qkv_dw<<<6912, 256, 0, stream>>>(q1, k1, v1, dwq, dwk, dwv, q2, k2, v2);
  k_knorm<<<32, 256, 0, stream>>>(k2, knorm);
  k_qv_pack<<<144, 256, 0, stream>>>(q2, v2, knorm, qn, vh);
  k_attn<<<2304, 256, SM_BYTES, stream>>>(qn, k2, vh, temp, aout);
  k_out<<<72, 256, 0, stream>>>(x, wo, aout, out);
}
